// VAE_42202348650518
// MI455X (gfx1250) — matrix operand order repaired and checked by reading, not compiled, not run on hardware
//
#include <hip/hip_runtime.h>
#include <hip/hip_bf16.h>

// ---------------------------------------------------------------------------
// VAE lower-bound, MI455X (gfx1250, wave32, WMMA).
//   N=16384, M=4096, K=64.  Memory-bound: ~512MB HBM traffic (x streamed 2x).
//   GEMMs via v_wmma_f32_16x16x32_bf16 (f32 accumulate).
// ---------------------------------------------------------------------------

typedef __attribute__((ext_vector_type(16))) __bf16 v16bf;
typedef __attribute__((ext_vector_type(8)))  float  v8f;
typedef __attribute__((ext_vector_type(4)))  float  v4f;

#define NROWS   16384
#define MDIM    4096
#define KDIM    64
#define LOG2PI  1.8378770664093453f

// ---- wave32 cross-lane helpers (explicit ds_bpermute, no wave64 assumptions)
__device__ __forceinline__ float wshfl(float v, int srcLane) {
    int r = __builtin_amdgcn_ds_bpermute(srcLane << 2, __builtin_bit_cast(int, v));
    return __builtin_bit_cast(float, r);
}
__device__ __forceinline__ float wshfl_xor(float v, int mask) {
    int lane = (int)(threadIdx.x & 31u);
    int r = __builtin_amdgcn_ds_bpermute(((lane ^ mask) & 31) << 2,
                                         __builtin_bit_cast(int, v));
    return __builtin_bit_cast(float, r);
}

__device__ __forceinline__ float dot4(v4f a, v4f b) {
    return a.x * b.x + a.y * b.y + a.z * b.z + a.w * b.w;
}

// Build a 16x32 bf16 A-fragment lane vector from 16 f32 values.
// ISA 16-bit A layout: elements 0..7 -> first 8-K chunk, 8..15 -> second chunk.
__device__ __forceinline__ v16bf to_bf16_frag(v4f a0, v4f a1, v4f a2, v4f a3) {
    v16bf f;
    f[0]  = (__bf16)a0.x; f[1]  = (__bf16)a0.y; f[2]  = (__bf16)a0.z; f[3]  = (__bf16)a0.w;
    f[4]  = (__bf16)a1.x; f[5]  = (__bf16)a1.y; f[6]  = (__bf16)a1.z; f[7]  = (__bf16)a1.w;
    f[8]  = (__bf16)a2.x; f[9]  = (__bf16)a2.y; f[10] = (__bf16)a2.z; f[11] = (__bf16)a2.w;
    f[12] = (__bf16)a3.x; f[13] = (__bf16)a3.y; f[14] = (__bf16)a3.z; f[15] = (__bf16)a3.w;
    return f;
}

// ---------------------------------------------------------------------------
// Kernel 0: pack W_enc_mu [4096,64] and W_dec_mu [64,4096] into bf16 WMMA-B
// fragment layout: fragment = (ktile of 32) x (ntile of 16); per lane 16
// contiguous bf16 = its column's 16 K-values (lanes 0-15: K 0-7,16-23, col=lane;
// lanes 16-31: K 8-15,24-31, col=lane-16).
// ---------------------------------------------------------------------------
__global__ void __launch_bounds__(256) pack_weights(
    const float* __restrict__ Wenc, const float* __restrict__ Wdec,
    __bf16* __restrict__ encP, __bf16* __restrict__ decP)
{
    int t = blockIdx.x * blockDim.x + threadIdx.x;   // 0..32767
    if (t < 16384) {
        // encoder: 128 mtiles * 4 ntiles fragments
        int lane  = t & 31;
        int frag  = t >> 5;          // mtile*4 + ntile
        int ntile = frag & 3;
        int mtile = frag >> 2;
        int col   = ntile * 16 + (lane & 15);
        int kbase = mtile * 32 + (lane >> 4) * 8;
        __bf16* dst = encP + (size_t)t * 16;
#pragma unroll
        for (int e = 0; e < 16; ++e)
            dst[e] = (__bf16)Wenc[(size_t)(kbase + e + (e & 8)) * KDIM + col];
    } else {
        // decoder: 256 ntiles * 2 ktiles fragments (ntile-major)
        int u     = t - 16384;
        int lane  = u & 31;
        int frag  = u >> 5;          // ntile*2 + kt
        int kt    = frag & 1;
        int ntile = frag >> 1;
        int col   = ntile * 16 + (lane & 15);
        int kbase = kt * 32 + (lane >> 4) * 8;
        __bf16* dst = decP + (size_t)u * 16;
#pragma unroll
        for (int e = 0; e < 16; ++e)
            dst[e] = (__bf16)Wdec[(size_t)(kbase + e + (e & 8)) * MDIM + col];
    }
}

// ---------------------------------------------------------------------------
// Kernel 1: encoder. One wave = one 16-row tile of x.
//   z_mu tile via 128 k-steps of wmma 16x16x32 (4 N-tiles), s_enc dot fused
//   on the same f32 stream.  Epilogue: z, log_post - log_prior, s_dec, var_x.
// ---------------------------------------------------------------------------
__global__ void __launch_bounds__(128) enc_kernel(
    const float* __restrict__ x, const float* __restrict__ noise,
    const __bf16* __restrict__ encP,
    const float* __restrict__ b_enc_mu, const float* __restrict__ W_enc_sigma,
    const float* __restrict__ b_enc_sigma,
    const float* __restrict__ W_dec_sigma, const float* __restrict__ b_dec_sigma,
    float* __restrict__ z_out, float* __restrict__ partial_out,
    float* __restrict__ varx_out)
{
    const int lane = threadIdx.x & 31;
    const int wave = threadIdx.x >> 5;
    const int tile = blockIdx.x * 4 + wave;      // 0..1023
    const int half = lane >> 4;                  // 0 / 1
    const int lrow = lane & 15;
    const float* xrow = x + (size_t)(tile * 16 + lrow) * MDIM;

    v8f acc[4] = {};
    float sacc = 0.f;

    for (int ks = 0; ks < 128; ++ks) {
        const int mb = ks * 32 + half * 8;
        // A: this lane's row, K chunks [mb..mb+7] and [mb+16..mb+23]
        v4f a0 = *(const v4f*)(xrow + mb);
        v4f a1 = *(const v4f*)(xrow + mb + 4);
        v4f a2 = *(const v4f*)(xrow + mb + 16);
        v4f a3 = *(const v4f*)(xrow + mb + 20);
        // sigma-head dot rides the same f32 data
        v4f s0 = *(const v4f*)(W_enc_sigma + mb);
        v4f s1 = *(const v4f*)(W_enc_sigma + mb + 4);
        v4f s2v = *(const v4f*)(W_enc_sigma + mb + 16);
        v4f s3 = *(const v4f*)(W_enc_sigma + mb + 20);
        sacc += dot4(a0, s0) + dot4(a1, s1) + dot4(a2, s2v) + dot4(a3, s3);

        v16bf af = to_bf16_frag(a0, a1, a2, a3);
        const __bf16* bp = encP + ((size_t)(ks * 4) * 32 + lane) * 16;
#pragma unroll
        for (int t = 0; t < 4; ++t) {
            v16bf bf = *(const v16bf*)(bp + (size_t)t * 512);
            acc[t] = __builtin_amdgcn_wmma_f32_16x16x32_bf16(
                false, af, false, bf, (short)0, acc[t], false, false);
        }
        // prefetch next k-step of the x stream
        if (ks + 1 < 128) __builtin_prefetch(xrow + mb + 32, 0, 3);
    }

    // s_enc per row: lane pair (l, l^16) holds disjoint halves of row l%16
    sacc += wshfl_xor(sacc, 16);
    const float s_enc = sacc + b_enc_sigma[0];
    const float s2    = s_enc * s_enc;           // reparam scale (sigma^2)
    const float bds   = b_dec_sigma[0];

#pragma unroll
    for (int r = 0; r < 8; ++r) {
        // accumulator row r maps to tile-row (8*half + r); fetch its sigma^2
        const float s2r   = wshfl(s2, r + 8 * half);
        const float varzr = s2r * s2r;           // sigma^4
        const size_t grow = (size_t)(tile * 16 + 8 * half + r);

        float zsq = 0.f, zsse = 0.f, zsd = 0.f;
#pragma unroll
        for (int t = 0; t < 4; ++t) {
            const int col = t * 16 + lrow;
            float zmu = acc[t][r] + b_enc_mu[col];
            float nz  = noise[grow * KDIM + col];
            float zv  = zmu + s2r * nz;
            z_out[grow * KDIM + col] = zv;
            zsq  += zv * zv;
            float d = zv - zmu;
            zsse += d * d;
            zsd  += zv * W_dec_sigma[col];
        }
        // reduce each row stat across the 16 lanes of this half
#pragma unroll
        for (int m = 1; m <= 8; m <<= 1) {
            zsq  += wshfl_xor(zsq,  m);
            zsse += wshfl_xor(zsse, m);
            zsd  += wshfl_xor(zsd,  m);
        }
        float sdr   = zsd + bds;
        float varx  = (sdr * sdr) * (sdr * sdr);
        float lpost = -0.5f * (KDIM * LOG2PI + KDIM * __logf(varzr) + zsse / varzr);
        float lpri  = -0.5f * (KDIM * LOG2PI + zsq);
        if (lrow == r) {
            partial_out[grow] = lpost - lpri;
            varx_out[grow]    = varx;
        }
    }
}

// ---------------------------------------------------------------------------
// Kernel 2: decoder + likelihood. One wave = one 16-row tile.  x_mu never
// touches memory: residual (x - x_mu)^2 fused into the WMMA epilogue while
// x streams through a second time.
// ---------------------------------------------------------------------------
__global__ void __launch_bounds__(128) dec_kernel(
    const float* __restrict__ x, const __bf16* __restrict__ decP,
    const float* __restrict__ b_dec_mu,
    const float* __restrict__ z_ws, const float* __restrict__ partial_ws,
    const float* __restrict__ varx_ws, float* __restrict__ tile_sums)
{
    const int lane = threadIdx.x & 31;
    const int wave = threadIdx.x >> 5;
    const int tile = blockIdx.x * 4 + wave;
    const int half = lane >> 4;
    const int lrow = lane & 15;

    // A = z row of this lane: two 16x32 bf16 fragments (K = 64)
    const float* zr = z_ws + (size_t)(tile * 16 + lrow) * KDIM;
    v16bf af[2];
#pragma unroll
    for (int c = 0; c < 2; ++c) {
        const float* p = zr + c * 32 + half * 8;
        af[c] = to_bf16_frag(*(const v4f*)(p),      *(const v4f*)(p + 4),
                             *(const v4f*)(p + 16), *(const v4f*)(p + 20));
    }

    float sse[8] = {0.f, 0.f, 0.f, 0.f, 0.f, 0.f, 0.f, 0.f};

    for (int nt = 0; nt < 256; ++nt) {
        const __bf16* bp = decP + (size_t)(nt * 2) * 512 + (size_t)lane * 16;
        v16bf b0 = *(const v16bf*)(bp);
        v16bf b1 = *(const v16bf*)(bp + 512);
        v8f c = {};
        c = __builtin_amdgcn_wmma_f32_16x16x32_bf16(false, af[0], false, b0,
                                                    (short)0, c, false, false);
        c = __builtin_amdgcn_wmma_f32_16x16x32_bf16(false, af[1], false, b1,
                                                    (short)0, c, false, false);
        const float bb = b_dec_mu[nt * 16 + lrow];
#pragma unroll
        for (int r = 0; r < 8; ++r) {
            const size_t grow = (size_t)(tile * 16 + 8 * half + r);
            float xv = x[grow * MDIM + nt * 16 + lrow];
            float d  = xv - (c[r] + bb);
            sse[r] += d * d;
        }
    }

    float tsum = 0.f;
#pragma unroll
    for (int r = 0; r < 8; ++r) {
        float s = sse[r];
#pragma unroll
        for (int m = 1; m <= 8; m <<= 1) s += wshfl_xor(s, m);
        const size_t grow = (size_t)(tile * 16 + 8 * half + r);
        float vx = varx_ws[grow];
        float ll = -0.5f * (MDIM * LOG2PI + MDIM * __logf(vx) + s / vx);
        tsum += partial_ws[grow] - ll;   // (log_post - log_prior) - log_like
    }
    tsum += wshfl_xor(tsum, 16);         // combine the two half-wave row sets
    if (lane == 0) tile_sums[tile] = tsum;
}

// ---------------------------------------------------------------------------
// Kernel 3: deterministic final reduction of 1024 tile partials -> scalar.
// ---------------------------------------------------------------------------
__global__ void __launch_bounds__(256) final_reduce(
    const float* __restrict__ tile_sums, float* __restrict__ out)
{
    __shared__ float ls[8];
    const int tid  = threadIdx.x;
    const int lane = tid & 31;
    const int wid  = tid >> 5;
    float s = 0.f;
    for (int i = tid; i < 1024; i += 256) s += tile_sums[i];
#pragma unroll
    for (int m = 16; m >= 1; m >>= 1) s += wshfl_xor(s, m);
    if (lane == 0) ls[wid] = s;
    __syncthreads();
    if (tid == 0) {
        float t = 0.f;
#pragma unroll
        for (int j = 0; j < 8; ++j) t += ls[j];
        out[0] = t;
    }
}

// ---------------------------------------------------------------------------
extern "C" void kernel_launch(void* const* d_in, const int* in_sizes, int n_in,
                              void* d_out, int out_size, void* d_ws, size_t ws_size,
                              hipStream_t stream) {
    const float* x       = (const float*)d_in[0];
    const float* noise   = (const float*)d_in[1];
    const float* Wenc    = (const float*)d_in[2];
    const float* b_enc_m = (const float*)d_in[3];
    const float* Wencsig = (const float*)d_in[4];
    const float* b_enc_s = (const float*)d_in[5];
    const float* Wdec    = (const float*)d_in[6];
    const float* b_dec_m = (const float*)d_in[7];
    const float* Wdecsig = (const float*)d_in[8];
    const float* b_dec_s = (const float*)d_in[9];

    // workspace layout
    float*  z_ws    = (float*)d_ws;                         // 16384*64 f32
    float*  part_ws = z_ws + (size_t)NROWS * KDIM;          // 16384 f32
    float*  varx_ws = part_ws + NROWS;                      // 16384 f32
    float*  tsums   = varx_ws + NROWS;                      // 1024 f32
    __bf16* encP    = (__bf16*)(tsums + 1024);              // 4096*64 bf16
    __bf16* decP    = encP + (size_t)MDIM * KDIM;           // 64*4096 bf16

    pack_weights<<<128, 256, 0, stream>>>(Wenc, Wdec, encP, decP);
    enc_kernel<<<256, 128, 0, stream>>>(x, noise, encP, b_enc_m, Wencsig,
                                        b_enc_s, Wdecsig, b_dec_s,
                                        z_ws, part_ws, varx_ws);
    dec_kernel<<<256, 128, 0, stream>>>(x, decP, b_dec_m, z_ws, part_ws,
                                        varx_ws, tsums);
    final_reduce<<<1, 256, 0, stream>>>(tsums, (float*)d_out);
}
